// GraphMaskedAttention_81080392613994
// MI455X (gfx1250) — matrix operand order repaired and checked by reading, not compiled, not run on hardware
//
#include <hip/hip_runtime.h>
#include <hip/hip_bf16.h>

typedef __attribute__((ext_vector_type(16))) __bf16 v16bf;
typedef __attribute__((ext_vector_type(8)))  float  v8f;

#define HDIM   512
#define NHEADS 8
#define HD     64

// ---------------- conversion kernels ----------------
__global__ void k_cvt_bf16(const float* __restrict__ in, __hip_bfloat16* __restrict__ out, int n) {
  int i = blockIdx.x * blockDim.x + threadIdx.x;
  if (i < n) out[i] = __float2bfloat16(in[i]);
}

// W [512,512] row-major f32 -> Wt bf16 with Wt[col*512 + k] = W[k*512 + col]
__global__ void k_cvt_w_t(const float* __restrict__ W, __hip_bfloat16* __restrict__ Wt) {
  int i = blockIdx.x * blockDim.x + threadIdx.x; // 512*512 threads
  int c = i >> 9, k = i & 511;
  Wt[(size_t)c * HDIM + k] = __float2bfloat16(W[(size_t)k * HDIM + c]);
}

__global__ void k_zero_u32(unsigned int* __restrict__ p, int n) {
  int i = blockIdx.x * blockDim.x + threadIdx.x;
  if (i < n) p[i] = 0u;
}

// boolean adjacency as bitmask: dedups duplicate edges like the reference .set(True)
__global__ void k_build_mask(const int* __restrict__ ei, int E, int nwords,
                             unsigned int* __restrict__ mask) {
  int e = blockIdx.x * blockDim.x + threadIdx.x;
  if (e < E) {
    int r = ei[e];        // edge_index[0][e] : query row
    int c = ei[E + e];    // edge_index[1][e] : key col
    atomicOr(&mask[(size_t)r * nwords + (c >> 5)], 1u << (c & 31));
  }
}

// ---------------- bf16 WMMA GEMM: C[M,512] = A[M,512] @ W[512,512] + bias ----------------
// A: bf16 row-major. Bt: bf16 column-major (Bt[n*512+k]). One wave -> 16x64 output strip.
union FragBF { unsigned int u[8]; v16bf v; };

__global__ __launch_bounds__(32) void k_gemm_bias(
    const __hip_bfloat16* __restrict__ A,
    const __hip_bfloat16* __restrict__ Bt,
    const float* __restrict__ bias,
    float* __restrict__ C)
{
  const int lane = threadIdx.x & 31;
  const int lo   = lane & 15;
  const int hi   = lane >> 4;
  const int row0 = blockIdx.x << 4;   // M tile
  const int col0 = blockIdx.y << 6;   // N strip of 64

  v8f acc[4] = {};

  const unsigned int* Arow = (const unsigned int*)(A + (size_t)(row0 + lo) * HDIM);
  const unsigned int* Bc[4];
#pragma unroll
  for (int t = 0; t < 4; ++t)
    Bc[t] = (const unsigned int*)(Bt + (size_t)(col0 + t * 16 + lo) * HDIM);

  for (int k0 = 0; k0 < HDIM; k0 += 32) {
    // A fragment, 16x32 bf16 (ISA 7.12.2): lane<16 -> K 0..7 & 16..23, lane>=16 -> K 8..15 & 24..31
    FragBF a;
#pragma unroll
    for (int p = 0; p < 8; ++p) {
      const int kof = ((p >= 4) ? 16 : 0) + hi * 8 + (p & 3) * 2;
      a.u[p] = Arow[(k0 + kof) >> 1];           // one 4B load = 2 contiguous bf16
    }
#pragma unroll
    for (int t = 0; t < 4; ++t) {
      // B fragment, 32x16 bf16: lane<16 -> K 0..7 & 16..23, lane>=16 -> K 8..15 & 24..31
      FragBF b;
#pragma unroll
      for (int p = 0; p < 8; ++p) {
        const int kof = ((p >= 4) ? 16 : 0) + hi * 8 + (p & 3) * 2;
        b.u[p] = Bc[t][(k0 + kof) >> 1];
      }
      acc[t] = __builtin_amdgcn_wmma_f32_16x16x32_bf16(
          /*neg_a=*/false, a.v, /*neg_b=*/false, b.v,
          /*c_mod=*/(short)0, acc[t], /*reuse_a=*/false, /*reuse_b=*/false);
    }
  }

  // C/D layout: VGPR r, lanes 0..15 -> M=r, lanes 16..31 -> M=8+r, N=lo
#pragma unroll
  for (int t = 0; t < 4; ++t) {
    const int col = col0 + t * 16 + lo;
    const float bval = bias[col];
#pragma unroll
    for (int r = 0; r < 8; ++r)
      C[(size_t)(row0 + hi * 8 + r) * HDIM + col] = acc[t][r] + bval;
  }
}

// ---------------- sparse masked attention: one wave per (row, head) ----------------
__global__ __launch_bounds__(32) void k_attn(
    const float* __restrict__ Q, const float* __restrict__ K, const float* __restrict__ V,
    const unsigned int* __restrict__ mask, float* __restrict__ att, int N)
{
  __shared__ int   s_cols[4096];
  __shared__ float s_sc[4096];
  __shared__ float s_q[HD];
  __shared__ int   s_cnt;

  const int row    = blockIdx.x;
  const int head   = blockIdx.y;
  const int lane   = threadIdx.x;
  const int nwords = N >> 5;

  if (lane == 0) s_cnt = 0;
  __syncthreads();

  // compact the bitmask row into an edge column list (order irrelevant for softmax)
  const unsigned int* mrow = mask + (size_t)row * nwords;
  for (int w = lane; w < nwords; w += 32) {
    unsigned int bits = mrow[w];
    while (bits) {
      int b = __builtin_ctz(bits);
      bits &= bits - 1u;
      int idx = atomicAdd(&s_cnt, 1);
      s_cols[idx] = (w << 5) + b;
    }
  }
  __syncthreads();
  const int n = s_cnt;

  float* orow = att + (size_t)row * HDIM + head * HD;
  if (n == 0) {                 // fully masked query row -> zero (reference has_edge path)
    orow[lane]      = 0.f;
    orow[lane + 32] = 0.f;
    return;
  }

  const size_t qoff = (size_t)row * HDIM + head * HD;
  s_q[lane]      = Q[qoff + lane];
  s_q[lane + 32] = Q[qoff + lane + 32];
  __syncthreads();

  // scores over edges only; scale = 1/sqrt(64)
  float lmax = -3.0e38f;
  for (int e = lane; e < n; e += 32) {
    const float* kp = K + (size_t)s_cols[e] * HDIM + head * HD;
    float dot = 0.f;
#pragma unroll
    for (int d = 0; d < HD; ++d) dot += s_q[d] * kp[d];
    dot *= 0.125f;
    s_sc[e] = dot;
    lmax = fmaxf(lmax, dot);
  }
#pragma unroll
  for (int m = 16; m; m >>= 1) lmax = fmaxf(lmax, __shfl_xor(lmax, m, 32));
  __syncthreads();

  float lsum = 0.f;
  for (int e = lane; e < n; e += 32) {
    float p = __expf(s_sc[e] - lmax);
    s_sc[e] = p;
    lsum += p;
  }
#pragma unroll
  for (int m = 16; m; m >>= 1) lsum += __shfl_xor(lsum, m, 32);
  __syncthreads();

  const float inv = 1.f / lsum;
  float a0 = 0.f, a1 = 0.f;
  for (int e = 0; e < n; ++e) {
    const float p = s_sc[e];
    const float* vp = V + (size_t)s_cols[e] * HDIM + head * HD;
    a0 += p * vp[lane];          // 64 floats contiguous across the wave per edge
    a1 += p * vp[lane + 32];
  }
  orow[lane]      = a0 * inv;
  orow[lane + 32] = a1 * inv;
}

// ---------------- host-side orchestration ----------------
extern "C" void kernel_launch(void* const* d_in, const int* in_sizes, int n_in,
                              void* d_out, int out_size, void* d_ws, size_t ws_size,
                              hipStream_t stream) {
  const float* x  = (const float*)d_in[0];
  const int*   ei = (const int*)d_in[1];
  const float* Wq = (const float*)d_in[2];
  const float* bq = (const float*)d_in[3];
  const float* Wk = (const float*)d_in[4];
  const float* bk = (const float*)d_in[5];
  const float* Wv = (const float*)d_in[6];
  const float* bv = (const float*)d_in[7];
  const float* Wo = (const float*)d_in[8];
  const float* bo = (const float*)d_in[9];
  float* out = (float*)d_out;

  const int H = HDIM;
  const int N = in_sizes[0] / H;   // 4096
  const int E = in_sizes[1] / 2;   // 131072
  const int nwords = N >> 5;

  char* ws = (char*)d_ws;
  size_t off = 0;
  float* Qf  = (float*)(ws + off); off += (size_t)N * H * sizeof(float);
  float* Kf  = (float*)(ws + off); off += (size_t)N * H * sizeof(float);
  float* Vf  = (float*)(ws + off); off += (size_t)N * H * sizeof(float);
  float* att = (float*)(ws + off); off += (size_t)N * H * sizeof(float);
  __hip_bfloat16* xb   = (__hip_bfloat16*)(ws + off); off += (size_t)N * H * 2;
  __hip_bfloat16* attb = (__hip_bfloat16*)(ws + off); off += (size_t)N * H * 2;
  __hip_bfloat16* Wqt  = (__hip_bfloat16*)(ws + off); off += (size_t)H * H * 2;
  __hip_bfloat16* Wkt  = (__hip_bfloat16*)(ws + off); off += (size_t)H * H * 2;
  __hip_bfloat16* Wvt  = (__hip_bfloat16*)(ws + off); off += (size_t)H * H * 2;
  __hip_bfloat16* Wot  = (__hip_bfloat16*)(ws + off); off += (size_t)H * H * 2;
  unsigned int* mask   = (unsigned int*)(ws + off);   off += (size_t)N * nwords * 4;

  const int nmask = N * nwords;
  k_zero_u32  <<<(nmask + 255) / 256, 256, 0, stream>>>(mask, nmask);
  k_build_mask<<<(E + 255) / 256,     256, 0, stream>>>(ei, E, nwords, mask);

  const int nx = N * H;
  k_cvt_bf16<<<(nx + 255) / 256, 256, 0, stream>>>(x, xb, nx);
  const int nw = H * H;
  k_cvt_w_t<<<(nw + 255) / 256, 256, 0, stream>>>(Wq, Wqt);
  k_cvt_w_t<<<(nw + 255) / 256, 256, 0, stream>>>(Wk, Wkt);
  k_cvt_w_t<<<(nw + 255) / 256, 256, 0, stream>>>(Wv, Wvt);
  k_cvt_w_t<<<(nw + 255) / 256, 256, 0, stream>>>(Wo, Wot);

  dim3 ggrid(N / 16, H / 64);
  k_gemm_bias<<<ggrid, 32, 0, stream>>>(xb, Wqt, bq, Qf);
  k_gemm_bias<<<ggrid, 32, 0, stream>>>(xb, Wkt, bk, Kf);
  k_gemm_bias<<<ggrid, 32, 0, stream>>>(xb, Wvt, bv, Vf);

  dim3 agrid(N, NHEADS);
  k_attn<<<agrid, 32, 0, stream>>>(Qf, Kf, Vf, mask, att, N);

  k_cvt_bf16<<<(nx + 255) / 256, 256, 0, stream>>>(att, attb, nx);
  k_gemm_bias<<<ggrid, 32, 0, stream>>>(attb, Wot, bo, out);
}
